// Multiheadattention_20229295964698
// MI455X (gfx1250) — matrix operand order repaired and checked by reading, not compiled, not run on hardware
//
#include <hip/hip_runtime.h>

// ---------------- types ----------------
typedef __attribute__((ext_vector_type(16))) __bf16 v16bf;
typedef __attribute__((ext_vector_type(8)))  __bf16 v8bf;
typedef __attribute__((ext_vector_type(4)))  __bf16 v4bf;
typedef __attribute__((ext_vector_type(8)))  float  v8f;
typedef __attribute__((ext_vector_type(4)))  unsigned int u32x4v;
typedef __attribute__((ext_vector_type(8)))  int          i32x8v;
typedef __attribute__((ext_vector_type(4)))  int          i32x4v;

#define HID   1024
#define SEQ   2048
#define BSZ   2
#define NH    16
#define HD    64
#define MTOT  (BSZ*SEQ)          // 4096 rows of activations

static __device__ __forceinline__ v16bf cat8(v8bf lo, v8bf hi) {
  return __builtin_shufflevector(lo, hi, 0,1,2,3,4,5,6,7,8,9,10,11,12,13,14,15);
}

// ---- Tensor Data Mover: 2-D tile load Global -> LDS (D# per ISA 08_async_tensor §8) ----
// tile_x elements contiguous (dim0), tile_y lines with stride0 elements; 2-byte elements.
static __device__ __forceinline__ void tdm_load_2d(
    unsigned int lds_byte_addr, const void* gptr,
    unsigned int tile_x, unsigned int tile_y,
    unsigned long long stride0_elems,
    unsigned int tensor_d0, unsigned int tensor_d1)
{
  unsigned long long ga = (unsigned long long)(uintptr_t)gptr;
  u32x4v g0;
  g0[0] = 1u;                                              // count=1, user descriptor
  g0[1] = lds_byte_addr;                                   // lds_addr [63:32]
  g0[2] = (unsigned int)ga;                                // global_addr[31:0]
  g0[3] = (unsigned int)((ga >> 32) & 0x01FFFFFFu) | (2u << 30); // addr[56:32], type=2
  i32x8v g1;
  g1[0] = 0x00010000;                                      // wg_mask=0, data_size=1 (2B)
  g1[1] = (int)((tensor_d0 & 0xFFFFu) << 16);              // tensor_dim0[15:0] @ [63:48]
  g1[2] = (int)(((tensor_d0 >> 16) & 0xFFFFu) | ((tensor_d1 & 0xFFFFu) << 16));
  g1[3] = (int)(((tensor_d1 >> 16) & 0xFFFFu) | ((tile_x & 0xFFFFu) << 16)); // tile_dim0
  g1[4] = (int)(tile_y & 0xFFFFu);                         // tile_dim1; tile_dim2=0
  g1[5] = (int)(stride0_elems & 0xFFFFFFFFull);            // tensor_dim0_stride[31:0]
  g1[6] = (int)((stride0_elems >> 32) & 0xFFFFull);        // stride[47:32]; dim1_stride=0
  g1[7] = 0;
  i32x4v z4 = {0, 0, 0, 0};
#if defined(__clang_major__) && (__clang_major__ >= 23)
  i32x8v z8 = {0, 0, 0, 0, 0, 0, 0, 0};
  __builtin_amdgcn_tensor_load_to_lds(g0, g1, z4, z4, z8, 0);
#else
  __builtin_amdgcn_tensor_load_to_lds(g0, g1, z4, z4, 0);
#endif
}

// ---------------- kernel 1: fp32 activations -> bf16 ----------------
__global__ __launch_bounds__(256) void cvt_act(
    const float* __restrict__ q, const float* __restrict__ k, const float* __restrict__ v,
    __bf16* __restrict__ qo, __bf16* __restrict__ ko, __bf16* __restrict__ vo)
{
  int z = blockIdx.z;
  const float* src = (z == 0) ? q : (z == 1) ? k : v;
  __bf16*      dst = (z == 0) ? qo : (z == 1) ? ko : vo;
  size_t i = ((size_t)blockIdx.x * 256 + threadIdx.x) * 4;
  float4 f = *reinterpret_cast<const float4*>(src + i);
  v4bf o;
  o.x = (__bf16)f.x; o.y = (__bf16)f.y; o.z = (__bf16)f.z; o.w = (__bf16)f.w;
  *reinterpret_cast<v4bf*>(dst + i) = o;
}

// ---------------- kernel 2: W[k][n] fp32 -> Wt[n][k] bf16 ----------------
__global__ __launch_bounds__(256) void transpose_w(
    const float* __restrict__ wq, const float* __restrict__ wk, const float* __restrict__ wv,
    __bf16* __restrict__ wqt, __bf16* __restrict__ wkt, __bf16* __restrict__ wvt)
{
  __shared__ float tile[32][33];
  int z = blockIdx.z;
  const float* W  = (z == 0) ? wq  : (z == 1) ? wk  : wv;
  __bf16*      Wt = (z == 0) ? wqt : (z == 1) ? wkt : wvt;
  int tx = threadIdx.x & 31;
  int ty = threadIdx.x >> 5;             // 0..7
  int k0 = blockIdx.x * 32, n0 = blockIdx.y * 32;
  #pragma unroll
  for (int i = 0; i < 32; i += 8)
    tile[ty + i][tx] = W[(size_t)(k0 + ty + i) * HID + n0 + tx];
  __syncthreads();
  #pragma unroll
  for (int i = 0; i < 32; i += 8)
    Wt[(size_t)(n0 + ty + i) * HID + k0 + tx] = (__bf16)tile[tx][ty + i];
}

// ---------------- kernel 3: fused QKV projection GEMM (bf16 WMMA) ----------------
// grid (4, 128, 3), block 256 (8 waves). Wave tile: 16(M) x 64(N) = one head wide.
__global__ __launch_bounds__(256) void qkv_gemm(
    const __bf16* __restrict__ qa, const __bf16* __restrict__ ka, const __bf16* __restrict__ va,
    const __bf16* __restrict__ wqt, const __bf16* __restrict__ wkt, const __bf16* __restrict__ wvt,
    const float* __restrict__ bq, const float* __restrict__ bk, const float* __restrict__ bv,
    __bf16* __restrict__ qp, __bf16* __restrict__ kp, __bf16* __restrict__ vtp)
{
  int z = blockIdx.z;
  const __bf16* A    = (z == 0) ? qa  : (z == 1) ? ka  : va;
  const __bf16* Wt   = (z == 0) ? wqt : (z == 1) ? wkt : wvt;
  const float*  bias = (z == 0) ? bq  : (z == 1) ? bk  : bv;

  int lane = threadIdx.x & 31;
  int wave = threadIdx.x >> 5;
  int lg   = lane & 15;
  int half = lane >> 4;
  int wm = wave & 1, wn = wave >> 1;
  int m0 = blockIdx.y * 32 + wm * 16;
  int n0 = blockIdx.x * 256 + wn * 64;     // multiple of 64 -> one head

  v8f acc[4] = {v8f{}, v8f{}, v8f{}, v8f{}};
  const __bf16* arow = A + (size_t)(m0 + lg) * HID;
  const __bf16* w0 = Wt + (size_t)(n0 +  0 + lg) * HID;
  const __bf16* w1 = Wt + (size_t)(n0 + 16 + lg) * HID;
  const __bf16* w2 = Wt + (size_t)(n0 + 32 + lg) * HID;
  const __bf16* w3 = Wt + (size_t)(n0 + 48 + lg) * HID;

  for (int kb = 0; kb < HID; kb += 32) {
    // A fragment: 16x32, lane = M row, elem j -> K = j + 16*half (contiguous)
    v8bf alo = *reinterpret_cast<const v8bf*>(arow + kb + 16 * half);
    v8bf ahi = *reinterpret_cast<const v8bf*>(arow + kb + 16 * half + 8);
    v16bf af = cat8(alo, ahi);
    // B fragments: 32x16, lane = N col, elem j -> K = j + 16*half (contiguous in Wt)
    v16bf b0 = *reinterpret_cast<const v16bf*>(w0 + kb + 16 * half);
    v16bf b1 = *reinterpret_cast<const v16bf*>(w1 + kb + 16 * half);
    v16bf b2 = *reinterpret_cast<const v16bf*>(w2 + kb + 16 * half);
    v16bf b3 = *reinterpret_cast<const v16bf*>(w3 + kb + 16 * half);
    acc[0] = __builtin_amdgcn_wmma_f32_16x16x32_bf16(false, af, false, b0, (short)0, acc[0], false, false);
    acc[1] = __builtin_amdgcn_wmma_f32_16x16x32_bf16(false, af, false, b1, (short)0, acc[1], false, false);
    acc[2] = __builtin_amdgcn_wmma_f32_16x16x32_bf16(false, af, false, b2, (short)0, acc[2], false, false);
    acc[3] = __builtin_amdgcn_wmma_f32_16x16x32_bf16(false, af, false, b3, (short)0, acc[3], false, false);
  }

  int bb = m0 >> 11;          // batch
  int srow = m0 & (SEQ - 1);  // seq row base
  int h = n0 >> 6;            // head
  if (z < 2) {
    // q/k: [b,h,s,d] bf16
    __bf16* outp = ((z == 0) ? qp : kp) + (size_t)(bb * NH + h) * SEQ * HD;
    #pragma unroll
    for (int i = 0; i < 4; i++) {
      int d = i * 16 + lg;
      float bvv = bias[n0 + i * 16 + lg];
      #pragma unroll
      for (int r = 0; r < 8; r++) {
        int s = srow + r + 8 * half;
        outp[(size_t)s * HD + d] = (__bf16)(acc[i][r] + bvv);
      }
    }
  } else {
    // v: transposed [b,h,d,s] bf16 so attention PV B-frags are contiguous over keys
    __bf16* outp = vtp + (size_t)(bb * NH + h) * HD * SEQ;
    #pragma unroll
    for (int i = 0; i < 4; i++) {
      int d = i * 16 + lg;
      float bvv = bias[n0 + i * 16 + lg];
      #pragma unroll
      for (int r = 0; r < 8; r++) {
        int s = srow + r + 8 * half;
        outp[(size_t)d * SEQ + s] = (__bf16)(acc[i][r] + bvv);
      }
    }
  }
}

// ---------------- kernel 4: flash attention, TDM-staged K/V, bf16 WMMA ----------------
// grid (SEQ/128, BSZ*NH), block 256 (8 waves). Each wave: one 16-query tile.
// K/V tiles for each 32-key chunk are DMA'd once per block into double-buffered LDS
// by the Tensor Data Mover (wave 0 issues, TENSORcnt + barriers synchronize).
__global__ __launch_bounds__(256) void attn(
    const __bf16* __restrict__ qp, const __bf16* __restrict__ kp, const __bf16* __restrict__ vtp,
    const float* __restrict__ maskp, float* __restrict__ out)
{
  __shared__ __align__(16) __bf16 Kt[2][32][64];            // 2 x 4KB, [key][d]
  __shared__ __align__(16) __bf16 Vt[2][64][32];            // 2 x 4KB, [d][key]
  __shared__ __align__(16) unsigned short Plds[8][16][40];  // per-wave P tile, padded

  int lane = threadIdx.x & 31;
  int wave = threadIdx.x >> 5;
  int lg   = lane & 15;
  int half = lane >> 4;
  int bh = blockIdx.y;            // b*NH + h
  int bb = bh >> 4;
  int hh = bh & 15;
  int qbase = blockIdx.x * 128 + wave * 16;

  const __bf16* qhead = qp  + (size_t)bh * SEQ * HD;
  const __bf16* khead = kp  + (size_t)bh * SEQ * HD;
  const __bf16* vhead = vtp + (size_t)bh * HD * SEQ;
  const float*  mrow  = maskp + (size_t)bb * SEQ;

  // load Q fragments once: two 16x32 A-frags covering d=0..63
  const __bf16* qrow = qhead + (size_t)(qbase + lg) * HD;
  v16bf qf[2];
  #pragma unroll
  for (int c = 0; c < 2; c++) {
    v8bf lo = *reinterpret_cast<const v8bf*>(qrow + 32 * c + 16 * half);
    v8bf hi = *reinterpret_cast<const v8bf*>(qrow + 32 * c + 16 * half + 8);
    qf[c] = cat8(lo, hi);
  }

  v8f acc[4] = {v8f{}, v8f{}, v8f{}, v8f{}};
  float mrun[8], lrun[8];
  #pragma unroll
  for (int r = 0; r < 8; r++) { mrun[r] = -3.0e38f; lrun[r] = 0.0f; }

  const int NCH = SEQ / 32;
  // prologue: DMA chunk 0 into buffer 0 (K: 64x32 tile, V: 32x64 tile)
  if (wave == 0) {
    tdm_load_2d((unsigned int)(uintptr_t)&Kt[0][0][0], khead, 64u, 32u, 64ull, 64u, SEQ);
    tdm_load_2d((unsigned int)(uintptr_t)&Vt[0][0][0], vhead, 32u, 64u, (unsigned long long)SEQ, SEQ, 64u);
  }

  for (int i = 0; i < NCH; i++) {
    int kc  = i * 32;
    int buf = i & 1;
    __syncthreads();   // everyone done reading buf^1 before TDM overwrites it
    if (wave == 0) {
      if (i + 1 < NCH) {
        tdm_load_2d((unsigned int)(uintptr_t)&Kt[buf ^ 1][0][0],
                    khead + (size_t)(kc + 32) * HD, 64u, 32u, 64ull, 64u, SEQ);
        tdm_load_2d((unsigned int)(uintptr_t)&Vt[buf ^ 1][0][0],
                    vhead + (kc + 32), 32u, 64u, (unsigned long long)SEQ, SEQ, 64u);
        __builtin_amdgcn_s_wait_tensorcnt((short)2);  // in-order: current buf's pair done
      } else {
        __builtin_amdgcn_s_wait_tensorcnt((short)0);
      }
    }
    __syncthreads();   // buf[cur] visible to all waves

    // ---- S = Q K^T for 32 keys (two 16x16 C tiles), K frags from LDS ----
    v8f s0 = {}, s1 = {};
    #pragma unroll
    for (int c = 0; c < 2; c++) {
      v16bf k0 = *reinterpret_cast<const v16bf*>(&Kt[buf][lg][32 * c + 16 * half]);
      v16bf k1 = *reinterpret_cast<const v16bf*>(&Kt[buf][16 + lg][32 * c + 16 * half]);
      s0 = __builtin_amdgcn_wmma_f32_16x16x32_bf16(false, qf[c], false, k0, (short)0, s0, false, false);
      s1 = __builtin_amdgcn_wmma_f32_16x16x32_bf16(false, qf[c], false, k1, (short)0, s1, false, false);
    }
    float mk0 = mrow[kc + lg];
    float mk1 = mrow[kc + 16 + lg];

    // ---- online softmax (row = r + 8*half, 16 cols across the 16-lane group) ----
    #pragma unroll
    for (int r = 0; r < 8; r++) {
      float x0 = (s0[r] + mk0) * 0.125f;    // mask added BEFORE 1/sqrt(64), as in reference
      float x1 = (s1[r] + mk1) * 0.125f;
      float mx = fmaxf(x0, x1);
      #pragma unroll
      for (int off = 8; off >= 1; off >>= 1) mx = fmaxf(mx, __shfl_xor(mx, off, 32));
      float mnew  = fmaxf(mrun[r], mx);
      float alpha = __expf(mrun[r] - mnew);
      float p0 = __expf(x0 - mnew);
      float p1 = __expf(x1 - mnew);
      float rs = p0 + p1;
      #pragma unroll
      for (int off = 8; off >= 1; off >>= 1) rs += __shfl_xor(rs, off, 32);
      lrun[r] = lrun[r] * alpha + rs;
      mrun[r] = mnew;
      acc[0][r] *= alpha; acc[1][r] *= alpha; acc[2][r] *= alpha; acc[3][r] *= alpha;
      // stage P (C-layout -> LDS) for transpose into A-layout
      Plds[wave][r + 8 * half][lg]      = __builtin_bit_cast(unsigned short, (__bf16)p0);
      Plds[wave][r + 8 * half][16 + lg] = __builtin_bit_cast(unsigned short, (__bf16)p1);
    }
    // intra-wave cross-lane LDS dependency: drain DS writes before transposed read
    asm volatile("s_wait_dscnt 0" ::: "memory");
    __builtin_amdgcn_wave_barrier();

    // ---- read P as 16x32 A-fragment: lane = query row, elems over the 32 keys ----
    const unsigned short* prow = &Plds[wave][lg][0];
    v8bf plo = *reinterpret_cast<const v8bf*>(prow + 16 * half);
    v8bf phi = *reinterpret_cast<const v8bf*>(prow + 16 * half + 8);
    v16bf pf = cat8(plo, phi);

    // ---- acc += P * V, V B-frags from LDS ([d][key] layout, contiguous over keys) ----
    #pragma unroll
    for (int nt = 0; nt < 4; nt++) {
      v16bf vf = *reinterpret_cast<const v16bf*>(&Vt[buf][nt * 16 + lg][16 * half]);
      acc[nt] = __builtin_amdgcn_wmma_f32_16x16x32_bf16(false, pf, false, vf, (short)0, acc[nt], false, false);
    }
    asm volatile("" ::: "memory");
  }

  // ---- epilogue: out[b, q, h*64+d] = acc / l ----
  #pragma unroll
  for (int nt = 0; nt < 4; nt++) {
    int d = nt * 16 + lg;
    #pragma unroll
    for (int r = 0; r < 8; r++) {
      int qi = qbase + r + 8 * half;
      out[((size_t)bb * SEQ + qi) * HID + hh * HD + d] = acc[nt][r] / lrun[r];
    }
  }
}

// ---------------- host launch ----------------
extern "C" void kernel_launch(void* const* d_in, const int* in_sizes, int n_in,
                              void* d_out, int out_size, void* d_ws, size_t ws_size,
                              hipStream_t stream) {
  const float* Q  = (const float*)d_in[0];
  const float* K  = (const float*)d_in[1];
  const float* V  = (const float*)d_in[2];
  const float* AM = (const float*)d_in[3];
  const float* Wq = (const float*)d_in[4];
  const float* bq = (const float*)d_in[5];
  const float* Wk = (const float*)d_in[6];
  const float* bk = (const float*)d_in[7];
  const float* Wv = (const float*)d_in[8];
  const float* bv = (const float*)d_in[9];
  float* out = (float*)d_out;

  const size_t ACT = (size_t)MTOT * HID;   // 4M elems
  const size_t WEL = (size_t)HID * HID;    // 1M elems
  __bf16* qa  = (__bf16*)d_ws;
  __bf16* ka  = qa  + ACT;
  __bf16* va  = ka  + ACT;
  __bf16* wqt = va  + ACT;
  __bf16* wkt = wqt + WEL;
  __bf16* wvt = wkt + WEL;
  __bf16* qp  = wvt + WEL;
  __bf16* kp  = qp  + ACT;
  __bf16* vtp = kp  + ACT;

  cvt_act<<<dim3((unsigned)(ACT / (256 * 4)), 1, 3), 256, 0, stream>>>(Q, K, V, qa, ka, va);
  transpose_w<<<dim3(HID / 32, HID / 32, 3), 256, 0, stream>>>(Wq, Wk, Wv, wqt, wkt, wvt);
  qkv_gemm<<<dim3(HID / 256, MTOT / 32, 3), 256, 0, stream>>>(
      qa, ka, va, wqt, wkt, wvt, bq, bk, bv, qp, kp, vtp);
  attn<<<dim3(SEQ / 128, BSZ * NH), 256, 0, stream>>>(qp, kp, vtp, AM, out);
}
